// SparseAttention_69887707841218
// MI455X (gfx1250) — matrix operand order repaired and checked by reading, not compiled, not run on hardware
//
#include <hip/hip_runtime.h>

#define BB   2
#define SS   2048
#define HIDN 1024
#define HH   16
#define DD   64
#define NQB  (SS / 64)      // 32 query blocks of 64
#define SCL  0.125f         // D^-0.5

typedef __attribute__((ext_vector_type(16))) _Float16 v16h;
typedef __attribute__((ext_vector_type(8)))  _Float16 v8h;
typedef __attribute__((ext_vector_type(8)))  float    v8f;
typedef __attribute__((ext_vector_type(4)))  unsigned int u32x4;
typedef __attribute__((ext_vector_type(8)))  int i32x8;
typedef __attribute__((ext_vector_type(4)))  int i32x4;

static __device__ __forceinline__ v16h cat8(v8h a, v8h b) {
  v16h r;
#pragma unroll
  for (int i = 0; i < 8; ++i) { r[i] = a[i]; r[i + 8] = b[i]; }
  return r;
}

// A fragment (row-major source), contiguous K order matching B:
// lanes 0-15: K {k0+0..15}; lanes 16-31: K {k0+16..31}.
static __device__ __forceinline__ v16h load_a_frag(const _Float16* rowptr, int k0, int hw) {
  const v8h* lo = (const v8h*)(rowptr + k0 + hw * 16);
  const v8h* hi = (const v8h*)(rowptr + k0 + hw * 16 + 8);
  return cat8(*lo, *hi);
}

// ---- Tensor Data Mover: 2D tile (rows x cols f16 elements) global -> LDS.
// D# per CDNA5 ISA. data_size=1 (2 bytes/elem).
static __device__ __forceinline__ void tdm_load_2d(void* lds, const void* gptr,
                                                   unsigned rows, unsigned cols,
                                                   unsigned stride_elems) {
  unsigned long long ga = (unsigned long long)(size_t)gptr;
  u32x4 g0;
  g0[0] = 1u;                                   // count=1 (valid user descriptor)
  g0[1] = (unsigned)(size_t)lds;                // lds_addr (byte offset)
  g0[2] = (unsigned)ga;                         // global_addr[31:0]
  g0[3] = ((unsigned)(ga >> 32) & 0x01FFFFFFu) | (2u << 30);  // addr[56:32] | type=2
  i32x8 g1;
  g1[0] = (int)(1u << 16);                      // data_size=1 -> 2 bytes
  g1[1] = (int)((cols & 0xFFFFu) << 16);        // tensor_dim0[15:0]
  g1[2] = (int)((cols >> 16) | ((rows & 0xFFFFu) << 16));   // tdim0 hi | tdim1 lo
  g1[3] = (int)((rows >> 16) | ((cols & 0xFFFFu) << 16));   // tdim1 hi | tile_dim0
  g1[4] = (int)(rows & 0xFFFFu);                // tile_dim1 | tile_dim2=0
  g1[5] = (int)stride_elems;                    // tensor_dim0_stride[31:0]
  g1[6] = 0;                                    // stride hi | dim1_stride lo
  g1[7] = 0;
  i32x4 z4 = (i32x4){0, 0, 0, 0};
#if __clang_major__ >= 23
  i32x8 z8 = (i32x8){0, 0, 0, 0, 0, 0, 0, 0};
  __builtin_amdgcn_tensor_load_to_lds(g0, g1, z4, z4, z8, 0);
#else
  __builtin_amdgcn_tensor_load_to_lds(g0, g1, z4, z4, 0);
#endif
}

// ---------------------------------------------------------------- convert
__global__ __launch_bounds__(256) void cvt_f32_f16(const float* __restrict__ s,
                                                   _Float16* __restrict__ d, int n) {
  for (int i = blockIdx.x * blockDim.x + threadIdx.x; i < n; i += gridDim.x * blockDim.x)
    d[i] = (_Float16)s[i];
}

// ---------------------------------------------------------------- GEMM core
// Workgroup = 4 waves; X strip (16 x 1024) staged once into LDS via TDM and
// shared; each wave owns a 16x64 output tile. K-loop software-pipelined.
// Variadic so the STORE statement may contain commas.
#define GEMM_BODY(...)                                                            \
  __shared__ __align__(64) _Float16 Xs[16][HIDN];                                 \
  const int m0   = blockIdx.x * 16;                                               \
  const int tid  = threadIdx.x;                                                   \
  const int wave = tid >> 5, lane = tid & 31;                                     \
  const int col  = lane & 15, hw = lane >> 4;                                     \
  const int n0   = blockIdx.y * 256 + wave * 64;                                  \
  if (wave == 0) {                                                                \
    tdm_load_2d(&Xs[0][0], X + (size_t)m0 * HIDN, 16u, (unsigned)HIDN,            \
                (unsigned)HIDN);                                                  \
    __builtin_amdgcn_s_wait_tensorcnt(0);                                         \
  }                                                                               \
  __syncthreads();                                                                \
  const _Float16* arow = &Xs[col][0];                                             \
  const _Float16* br0 = W + (size_t)(n0 +  0 + col) * HIDN;                       \
  const _Float16* br1 = W + (size_t)(n0 + 16 + col) * HIDN;                       \
  const _Float16* br2 = W + (size_t)(n0 + 32 + col) * HIDN;                       \
  const _Float16* br3 = W + (size_t)(n0 + 48 + col) * HIDN;                       \
  v8f acc0 = {}, acc1 = {}, acc2 = {}, acc3 = {};                                 \
  v16h a  = load_a_frag(arow, 0, hw);                                             \
  v16h b0 = *(const v16h*)(br0 + hw * 16);                                        \
  v16h b1 = *(const v16h*)(br1 + hw * 16);                                        \
  v16h b2 = *(const v16h*)(br2 + hw * 16);                                        \
  v16h b3 = *(const v16h*)(br3 + hw * 16);                                        \
  _Pragma("unroll 4")                                                             \
  for (int k0 = 0; k0 < HIDN; k0 += 32) {                                         \
    v16h ac = a, c0 = b0, c1 = b1, c2 = b2, c3 = b3;                              \
    int kn = k0 + 32;                                                             \
    if (kn < HIDN) {   /* prefetch next k-step while WMMAs run */                 \
      __builtin_prefetch((const void*)(br0 + kn + 32), 0, 3);                     \
      a  = load_a_frag(arow, kn, hw);                                             \
      b0 = *(const v16h*)(br0 + kn + hw * 16);                                    \
      b1 = *(const v16h*)(br1 + kn + hw * 16);                                    \
      b2 = *(const v16h*)(br2 + kn + hw * 16);                                    \
      b3 = *(const v16h*)(br3 + kn + hw * 16);                                    \
    }                                                                             \
    acc0 = __builtin_amdgcn_wmma_f32_16x16x32_f16(false, ac, false, c0, (short)0, acc0, false, false); \
    acc1 = __builtin_amdgcn_wmma_f32_16x16x32_f16(false, ac, false, c1, (short)0, acc1, false, false); \
    acc2 = __builtin_amdgcn_wmma_f32_16x16x32_f16(false, ac, false, c2, (short)0, acc2, false, false); \
    acc3 = __builtin_amdgcn_wmma_f32_16x16x32_f16(false, ac, false, c3, (short)0, acc3, false, false); \
  }                                                                               \
  v8f* accs[4] = {&acc0, &acc1, &acc2, &acc3};                                    \
  _Pragma("unroll")                                                               \
  for (int t = 0; t < 4; ++t) {                                                   \
    _Pragma("unroll")                                                             \
    for (int r = 0; r < 8; ++r) {                                                 \
      int m = m0 + r + 8 * hw;                                                    \
      int n = n0 + t * 16 + col;                                                  \
      __VA_ARGS__;                                                                \
    }                                                                             \
  }

// QKV GEMM: f16 output scattered to [B,H,S,D]
__global__ __launch_bounds__(128) void gemm_qkv_f16(const _Float16* __restrict__ X,
                                                    const _Float16* __restrict__ W,
                                                    _Float16* __restrict__ Y) {
  GEMM_BODY(
    int bq = m >> 11; int sq = m & (SS - 1);
    int hq = n >> 6;  int dq = n & (DD - 1);
    Y[(((size_t)(bq * HH + hq) * SS + sq) * DD) + dq] = (_Float16)((*accs[t])[r])
  )
}

// Output projection: f32 row-major
__global__ __launch_bounds__(128) void gemm_out_f32(const _Float16* __restrict__ X,
                                                    const _Float16* __restrict__ W,
                                                    float* __restrict__ Y) {
  GEMM_BODY(
    Y[(size_t)m * HIDN + n] = (*accs[t])[r]
  )
}

// ---------------------------------------------------------------- sparse attention
// 1 workgroup (4 waves) per (b,h,64-query block). K tile staged via TDM,
// V tile transposed into LDS, P re-swizzled through LDS for the PV GEMM.
__global__ __launch_bounds__(128) void attn_sparse(const _Float16* __restrict__ Q,
                                                   const _Float16* __restrict__ K,
                                                   const _Float16* __restrict__ V,
                                                   _Float16* __restrict__ AO) {
  __shared__ __align__(64) _Float16 Kt[80][64];      // rows 0-63 local, 64-79 global
  __shared__ __align__(64) _Float16 Vt[DD][96];      // transposed V: [d][key]
  __shared__ __align__(64) _Float16 P[4][16][96];    // per-wave probability strip

  const int bid = blockIdx.x;
  const int qb  = bid % NQB;
  const int h   = (bid / NQB) % HH;
  const int b   = bid / (NQB * HH);

  const _Float16* Qh = Q + (size_t)(b * HH + h) * SS * DD;
  const _Float16* Kh = K + (size_t)(b * HH + h) * SS * DD;
  const _Float16* Vh = V + (size_t)(b * HH + h) * SS * DD;

  const int tid  = threadIdx.x;
  const int wave = tid >> 5;
  const int lane = tid & 31;
  const int col  = lane & 15;
  const int hw   = lane >> 4;
  const int m0   = wave * 16;

  // ---- TDM: stage K tile (64 local rows + 16 global rows), wave 0 only
  if (wave == 0) {
    tdm_load_2d(&Kt[0][0],  Kh + (size_t)(qb * 64) * DD, 64u, (unsigned)DD, (unsigned)DD);
    tdm_load_2d(&Kt[64][0], Kh,                          16u, (unsigned)DD, (unsigned)DD);
    __builtin_amdgcn_s_wait_tensorcnt(0);
  }

  // ---- stage V (80 keys x 64 d) transposed into LDS; keys 80..95 zeroed
  for (int c = tid; c < 640; c += 128) {           // uniform trip count
    int key = c >> 3, dg = c & 7;
    int srow = (key < 64) ? (qb * 64 + key) : (key - 64);
    v8h vv = *(const v8h*)(Vh + (size_t)srow * DD + dg * 8);
#pragma unroll
    for (int j = 0; j < 8; ++j) Vt[dg * 8 + j][key] = vv[j];
  }
  for (int c = tid; c < 1024; c += 128) Vt[c >> 4][80 + (c & 15)] = (_Float16)0.f;
  __syncthreads();

  // ---- scores: 5 N-tiles x 2 K-steps of WMMA (B fragments from LDS K tile)
  v16h qa0 = load_a_frag(Qh + (size_t)(qb * 64 + m0 + col) * DD, 0, hw);
  v16h qa1 = load_a_frag(Qh + (size_t)(qb * 64 + m0 + col) * DD, 32, hw);

  v8f acc[5];
#pragma unroll
  for (int t = 0; t < 5; ++t) acc[t] = (v8f){};
#pragma unroll
  for (int t = 0; t < 5; ++t) {
    const _Float16* kr = &Kt[t * 16 + col][0];     // t==4 -> rows 64..79 (global keys)
    v16h b0 = *(const v16h*)(kr + 0 + hw * 16);
    v16h b1 = *(const v16h*)(kr + 32 + hw * 16);
    acc[t] = __builtin_amdgcn_wmma_f32_16x16x32_f16(false, qa0, false, b0, (short)0, acc[t], false, false);
    acc[t] = __builtin_amdgcn_wmma_f32_16x16x32_f16(false, qa1, false, b1, (short)0, acc[t], false, false);
  }

  // ---- scale + sparse mask (C layout: row = r + 8*hw, col = lane&15)
  const float NEG = -__builtin_inff();
#pragma unroll
  for (int t = 0; t < 5; ++t) {
#pragma unroll
    for (int r = 0; r < 8; ++r) {
      int qloc = m0 + r + 8 * hw;
      bool ok = (t < 4) ? ((t * 16 + col) <= qloc)   // causal within own block
                        : (qb > 0);                  // global cols valid iff block>0
      acc[t][r] = ok ? acc[t][r] * SCL : NEG;
    }
  }

  // ---- row softmax (reduce across the 16 lanes of this half)
  float mx[8], sm[8];
#pragma unroll
  for (int r = 0; r < 8; ++r) {
    float m = acc[0][r];
#pragma unroll
    for (int t = 1; t < 5; ++t) m = fmaxf(m, acc[t][r]);
    mx[r] = m;
  }
#pragma unroll
  for (int off = 8; off; off >>= 1)
#pragma unroll
    for (int r = 0; r < 8; ++r) mx[r] = fmaxf(mx[r], __shfl_xor(mx[r], off, 32));
#pragma unroll
  for (int r = 0; r < 8; ++r) sm[r] = 0.f;
#pragma unroll
  for (int t = 0; t < 5; ++t)
#pragma unroll
    for (int r = 0; r < 8; ++r) {
      float e = __expf(acc[t][r] - mx[r]);           // exp(-inf)=0 for masked
      acc[t][r] = e;
      sm[r] += e;
    }
#pragma unroll
  for (int off = 8; off; off >>= 1)
#pragma unroll
    for (int r = 0; r < 8; ++r) sm[r] += __shfl_xor(sm[r], off, 32);
  float inv[8];
#pragma unroll
  for (int r = 0; r < 8; ++r) inv[r] = 1.f / sm[r];

  // ---- write probabilities to LDS (C-layout -> A-layout), pad cols 80..95
#pragma unroll
  for (int t = 0; t < 5; ++t)
#pragma unroll
    for (int r = 0; r < 8; ++r)
      P[wave][r + 8 * hw][t * 16 + col] = (_Float16)(acc[t][r] * inv[r]);
  for (int c = lane; c < 256; c += 32) P[wave][c >> 4][80 + (c & 15)] = (_Float16)0.f;

  // ---- out = P @ V : 3 K-steps (96) x 4 N-tiles (64 d)
  v8f o[4];
#pragma unroll
  for (int t = 0; t < 4; ++t) o[t] = (v8f){};
#pragma unroll
  for (int ks = 0; ks < 3; ++ks) {
    int k0 = ks * 32;
    v16h a = load_a_frag(&P[wave][col][0], k0, hw);
#pragma unroll
    for (int t = 0; t < 4; ++t) {
      v16h bf = *(const v16h*)(&Vt[t * 16 + col][0] + k0 + hw * 16);
      o[t] = __builtin_amdgcn_wmma_f32_16x16x32_f16(false, a, false, bf, (short)0, o[t], false, false);
    }
  }

  // ---- store attention output token-major [B*S, HID] f16
#pragma unroll
  for (int t = 0; t < 4; ++t)
#pragma unroll
    for (int r = 0; r < 8; ++r) {
      int srow = qb * 64 + m0 + r + 8 * hw;
      AO[((size_t)(b * SS + srow)) * HIDN + h * DD + t * 16 + col] = (_Float16)o[t][r];
    }
}

// ---------------------------------------------------------------- launch
extern "C" void kernel_launch(void* const* d_in, const int* in_sizes, int n_in,
                              void* d_out, int out_size, void* d_ws, size_t ws_size,
                              hipStream_t stream) {
  const float* hs = (const float*)d_in[0];
  const float* Wq = (const float*)d_in[1];
  const float* Wk = (const float*)d_in[2];
  const float* Wv = (const float*)d_in[3];
  const float* Wo = (const float*)d_in[4];

  const size_t NX = (size_t)BB * SS * HIDN;   // 4,194,304
  const size_t NW = (size_t)HIDN * HIDN;      // 1,048,576

  _Float16* Xh  = (_Float16*)d_ws;
  _Float16* Wqh = Xh + NX;
  _Float16* Wkh = Wqh + NW;
  _Float16* Wvh = Wkh + NW;
  _Float16* Woh = Wvh + NW;
  _Float16* Qh  = Woh + NW;
  _Float16* Kh  = Qh + NX;
  _Float16* Vh  = Kh + NX;
  _Float16* AOh = Vh + NX;   // total ~50.4 MB

  cvt_f32_f16<<<1024, 256, 0, stream>>>(hs, Xh, (int)NX);
  cvt_f32_f16<<<512, 256, 0, stream>>>(Wq, Wqh, (int)NW);
  cvt_f32_f16<<<512, 256, 0, stream>>>(Wk, Wkh, (int)NW);
  cvt_f32_f16<<<512, 256, 0, stream>>>(Wv, Wvh, (int)NW);
  cvt_f32_f16<<<512, 256, 0, stream>>>(Wo, Woh, (int)NW);

  dim3 gg(256, 4);  // 4096/16 M-tiles x 1024/256 N-groups (4 waves x 64)
  gemm_qkv_f16<<<gg, 128, 0, stream>>>(Xh, Wqh, Qh);
  gemm_qkv_f16<<<gg, 128, 0, stream>>>(Xh, Wkh, Kh);
  gemm_qkv_f16<<<gg, 128, 0, stream>>>(Xh, Wvh, Vh);

  attn_sparse<<<BB * HH * NQB, 128, 0, stream>>>(Qh, Kh, Vh, AOh);

  gemm_out_f32<<<gg, 128, 0, stream>>>(AOh, Woh, (float*)d_out);
}
